// RowWiseExpertsMLP_21406117003383
// MI455X (gfx1250) — matrix operand order repaired and checked by reading, not compiled, not run on hardware
//
#include <hip/hip_runtime.h>
#include <hip/hip_bf16.h>

// ---------------------------------------------------------------------------
// MoE expert MLP: y_i = gelu(x_i @ W1[eid_i]^T) @ W2[eid_i]^T + b2[eid_i]
// Grouped-GEMM dispatch, bf16 WMMA (v_wmma_f32_16x16x32_bf16) + f32 accum,
// async global->LDS staging (ASYNCcnt) when the toolchain exposes it.
// ---------------------------------------------------------------------------

typedef __attribute__((ext_vector_type(16))) __bf16 v16bf;
typedef __attribute__((ext_vector_type(8)))  __bf16 v8bf;
typedef __attribute__((ext_vector_type(4)))  __bf16 v4bf;
typedef __attribute__((ext_vector_type(8)))  float  v8f;
typedef int v4i __attribute__((vector_size(4 * sizeof(int))));

#define AS1 __attribute__((address_space(1)))
#define AS3 __attribute__((address_space(3)))

#define D_DIM   1024
#define H1_DIM  4096
#define O_DIM   1024
#define E_NUM   8
#define B_TOK   8192
#define BM      128
#define BN      128
#define BK      64
#define STR     72            // LDS row stride (bf16): 144 B, 16B-aligned
#define MAX_TILES (B_TOK / BM + E_NUM)   // 72

#if __has_builtin(__builtin_amdgcn_global_load_async_to_lds_b128)
#define USE_ASYNC_LDS 1
#else
#define USE_ASYNC_LDS 0
#endif

__device__ __forceinline__ void wait_async_zero() {
#if __has_builtin(__builtin_amdgcn_s_wait_asynccnt)
    __builtin_amdgcn_s_wait_asynccnt(0);
#else
    asm volatile("s_wait_asynccnt 0x0" ::: "memory");
#endif
}

// ---------------------------- setup kernels --------------------------------

__global__ void k_init(int* meta) {
    if (threadIdx.x < 32) meta[threadIdx.x] = 0;
}

__global__ void k_count(const int* __restrict__ eid, int* __restrict__ meta,
                        int* __restrict__ pos) {
    int i = blockIdx.x * blockDim.x + threadIdx.x;
    if (i < B_TOK) {
        int e = eid[i];
        pos[i] = atomicAdd(&meta[e], 1);
    }
}

__global__ void k_scan(int* __restrict__ meta, int* __restrict__ tileExp,
                       int* __restrict__ tileRow) {
    if (blockIdx.x == 0 && threadIdx.x == 0) {
        int base = 0;
        int T = 0;
        meta[8] = 0;
        for (int e = 0; e < E_NUM; ++e) {
            int c  = meta[e];
            int nt = (c + BM - 1) / BM;
            for (int t = 0; t < nt; ++t) {
                tileExp[T] = e;
                tileRow[T] = base + t * BM;
                ++T;
            }
            base += c;
            meta[9 + e] = base;     // meta[8+e+1] = group end
        }
        meta[17] = T;               // total M-tiles
    }
}

__global__ void k_scatter(const int* __restrict__ eid, const int* __restrict__ meta,
                          const int* __restrict__ pos, int* __restrict__ order) {
    int i = blockIdx.x * blockDim.x + threadIdx.x;
    if (i < B_TOK) {
        int e = eid[i];
        order[meta[8 + e] + pos[i]] = i;
    }
}

// fp32 -> bf16 (RNE), 4 elems/thread, grid-stride
__global__ void k_cvt_bf16(const float* __restrict__ src, __bf16* __restrict__ dst,
                           long n) {
    long stride = (long)gridDim.x * blockDim.x * 4;
    for (long i = ((long)blockIdx.x * blockDim.x + threadIdx.x) * 4; i < n;
         i += stride) {
        float4 v = *(const float4*)(src + i);
        v4bf o;
        o.x = (__bf16)v.x; o.y = (__bf16)v.y;
        o.z = (__bf16)v.z; o.w = (__bf16)v.w;
        *(v4bf*)(dst + i) = o;
    }
}

// ------------------------------ grouped GEMM --------------------------------

__device__ __forceinline__ float gelu_f(float x) {
    return 0.5f * x * (1.0f + erff(x * 0.70710678118654752440f));
}

// C[m][n] = sum_k A[m][k] * W[e][n][k]  (+gelu / +bias+scatter)
template <int KDIM, bool IS_FIRST>
__global__ __launch_bounds__(256) void moe_gemm(
    const __bf16* __restrict__ A,      // IS_FIRST: xb [B,D] (gather by order) ; else hb grouped
    const __bf16* __restrict__ W,      // [E, Ndim, KDIM] bf16
    const float*  __restrict__ bias,   // !IS_FIRST: [E, Ndim]
    __bf16* __restrict__ outb,         // IS_FIRST: hb [B, Ndim] grouped rows
    float*  __restrict__ outf,         // !IS_FIRST: y [B, Ndim] scattered
    const int* __restrict__ meta,
    const int* __restrict__ tileExp,
    const int* __restrict__ tileRow,
    const int* __restrict__ order,
    int Ndim) {
    const int T = blockIdx.y;
    if (T >= meta[17]) return;

    const int e    = tileExp[T];
    const int grow = tileRow[T];
    int rows = meta[8 + e + 1] - grow;
    if (rows > BM) rows = BM;
    const int nblk = blockIdx.x * BN;

    __shared__ __bf16 xs[BM * STR];
    __shared__ __bf16 wl[BN * STR];

    const int t  = threadIdx.x;
    const int r  = t >> 1;     // staged row (0..127)
    const int hf = t & 1;      // 32-element half of the 64-wide K slab

    // A-row source pointer (gather for layer 1, direct grouped rows for layer 2)
    const bool avalid = (r < rows);
    const __bf16* arow;
    if (IS_FIRST) {
        int token = avalid ? order[grow + r] : 0;
        arow = A + (size_t)token * KDIM;
    } else {
        arow = A + (size_t)(grow + r) * KDIM;
    }
    const __bf16* wrow = W + ((size_t)e * Ndim + (nblk + r)) * (size_t)KDIM;

    // wave32 tiling: 8 waves -> 4 (M) x 2 (N); each wave = 32x64 = 2x4 WMMA tiles
    const int w    = t >> 5;
    const int ln   = t & 31;
    const int mw   = (w & 3) * 32;
    const int nw   = (w >> 2) * 64;
    const int half = ln >> 4;
    const int l16  = ln & 15;

    // Padded A-rows: zero once (async DMA can't synthesize zeros; these rows
    // are never overwritten afterwards).
    if (!avalid) {
        int4 z = {0, 0, 0, 0};
        int4* d = (int4*)(xs + r * STR + hf * 32);
        d[0] = z; d[1] = z; d[2] = z; d[3] = z;
    }

    v8f acc[2][4];
#pragma unroll
    for (int i = 0; i < 2; ++i)
#pragma unroll
        for (int j = 0; j < 4; ++j)
            acc[i][j] = (v8f){0.f, 0.f, 0.f, 0.f, 0.f, 0.f, 0.f, 0.f};

    for (int k0 = 0; k0 < KDIM; k0 += BK) {
        __syncthreads();   // previous compute done before restaging LDS
#if USE_ASYNC_LDS
        // ---- async DMA staging: 64 B per thread per tile, no VGPR round-trip
        if (avalid) {
#pragma unroll
            for (int c = 0; c < 4; ++c)
                __builtin_amdgcn_global_load_async_to_lds_b128(
                    (AS1 v4i*)(arow + k0 + hf * 32 + c * 8),
                    (AS3 v4i*)(xs + r * STR + hf * 32 + c * 8),
                    0, 0);
        }
#pragma unroll
        for (int c = 0; c < 4; ++c)
            __builtin_amdgcn_global_load_async_to_lds_b128(
                (AS1 v4i*)(wrow + k0 + hf * 32 + c * 8),
                (AS3 v4i*)(wl + r * STR + hf * 32 + c * 8),
                0, 0);
        wait_async_zero();
#else
        // ---- manual staging fallback ----
        {
            int4 q0 = {0,0,0,0}, q1 = {0,0,0,0}, q2 = {0,0,0,0}, q3 = {0,0,0,0};
            if (avalid) {
                const int4* p = (const int4*)(arow + k0 + hf * 32);
                q0 = p[0]; q1 = p[1]; q2 = p[2]; q3 = p[3];
            }
            int4* d = (int4*)(xs + r * STR + hf * 32);
            d[0] = q0; d[1] = q1; d[2] = q2; d[3] = q3;
        }
        {
            const int4* p = (const int4*)(wrow + k0 + hf * 32);
            int4 q0 = p[0], q1 = p[1], q2 = p[2], q3 = p[3];
            int4* d = (int4*)(wl + r * STR + hf * 32);
            d[0] = q0; d[1] = q1; d[2] = q2; d[3] = q3;
            if (k0 + BK < KDIM)
                __builtin_prefetch(wrow + k0 + BK + hf * 32, 0, 0);
        }
#endif
        __syncthreads();

        // ---- two K=32 sub-steps; hoist ALL fragment loads before the WMMAs
        // so the matrix pipe isn't gated on a full dscnt drain per tile.
#pragma unroll
        for (int ks = 0; ks < BK; ks += 32) {
            v16bf afr[2];
            v16bf bfr[4];
#pragma unroll
            for (int nt = 0; nt < 4; ++nt) {
                const __bf16* cp = wl + (nw + nt * 16 + l16) * STR + ks + half * 16;
                v8bf lo = *(const v8bf*)(cp);
                v8bf hi = *(const v8bf*)(cp + 8);
                bfr[nt] = __builtin_shufflevector(lo, hi, 0, 1, 2, 3, 4, 5, 6, 7,
                                                  8, 9, 10, 11, 12, 13, 14, 15);
            }
#pragma unroll
            for (int mt = 0; mt < 2; ++mt) {
                const __bf16* rp = xs + (mw + mt * 16 + l16) * STR + ks;
                v8bf lo = *(const v8bf*)(rp + half * 16);
                v8bf hi = *(const v8bf*)(rp + half * 16 + 8);
                afr[mt] = __builtin_shufflevector(lo, hi, 0, 1, 2, 3, 4, 5, 6, 7,
                                                  8, 9, 10, 11, 12, 13, 14, 15);
            }
#pragma unroll
            for (int nt = 0; nt < 4; ++nt)
#pragma unroll
                for (int mt = 0; mt < 2; ++mt)
                    acc[mt][nt] = __builtin_amdgcn_wmma_f32_16x16x32_bf16(
                        false, afr[mt], false, bfr[nt], (short)0, acc[mt][nt],
                        false, false);
        }
    }

    // ---- epilogue: C layout — VGPR v: M = half*8 + v, N = lane&15 ----
#pragma unroll
    for (int mt = 0; mt < 2; ++mt) {
#pragma unroll
        for (int v = 0; v < 8; ++v) {
            const int m_local = mw + mt * 16 + half * 8 + v;
            if (m_local < rows) {
                if (IS_FIRST) {
                    const size_t rowoff = (size_t)(grow + m_local) * Ndim + nblk;
#pragma unroll
                    for (int nt = 0; nt < 4; ++nt) {
                        float hv = gelu_f(acc[mt][nt][v]);
                        outb[rowoff + nw + nt * 16 + l16] = (__bf16)hv;
                    }
                } else {
                    const int token = order[grow + m_local];
                    const size_t rowoff = (size_t)token * Ndim;
#pragma unroll
                    for (int nt = 0; nt < 4; ++nt) {
                        const int col = nblk + nw + nt * 16 + l16;
                        outf[rowoff + col] =
                            acc[mt][nt][v] + bias[(size_t)e * Ndim + col];
                    }
                }
            }
        }
    }
}

// ------------------------------- launcher -----------------------------------

extern "C" void kernel_launch(void* const* d_in, const int* in_sizes, int n_in,
                              void* d_out, int out_size, void* d_ws,
                              size_t ws_size, hipStream_t stream) {
    (void)in_sizes; (void)n_in; (void)out_size; (void)ws_size;
    const float* x   = (const float*)d_in[0];   // [B, D]
    const int*   eid = (const int*)d_in[1];     // [B]
    const float* W1  = (const float*)d_in[2];   // [E, H1, D]
    const float* W2  = (const float*)d_in[3];   // [E, O, H1]
    const float* b2  = (const float*)d_in[4];   // [E, O]
    float* y = (float*)d_out;                   // [B, O]

    char*  ws  = (char*)d_ws;
    size_t off = 0;
    auto alloc = [&](size_t bytes) -> char* {
        char* p = ws + off;
        off += (bytes + 255) & ~(size_t)255;
        return p;
    };
    int* meta    = (int*)alloc(32 * sizeof(int));
    int* tileExp = (int*)alloc(MAX_TILES * sizeof(int));
    int* tileRow = (int*)alloc(MAX_TILES * sizeof(int));
    int* pos     = (int*)alloc((size_t)B_TOK * sizeof(int));
    int* order   = (int*)alloc((size_t)B_TOK * sizeof(int));
    __bf16* xb   = (__bf16*)alloc((size_t)B_TOK * D_DIM * 2);
    __bf16* w1b  = (__bf16*)alloc((size_t)E_NUM * H1_DIM * D_DIM * 2);
    __bf16* w2b  = (__bf16*)alloc((size_t)E_NUM * O_DIM * H1_DIM * 2);
    __bf16* hb   = (__bf16*)alloc((size_t)B_TOK * H1_DIM * 2);

    k_init<<<1, 32, 0, stream>>>(meta);
    k_cvt_bf16<<<2048, 256, 0, stream>>>(x, xb, (long)B_TOK * D_DIM);
    k_cvt_bf16<<<4096, 256, 0, stream>>>(W1, w1b, (long)E_NUM * H1_DIM * D_DIM);
    k_cvt_bf16<<<4096, 256, 0, stream>>>(W2, w2b, (long)E_NUM * O_DIM * H1_DIM);
    k_count<<<B_TOK / 256, 256, 0, stream>>>(eid, meta, pos);
    k_scan<<<1, 1, 0, stream>>>(meta, tileExp, tileRow);
    k_scatter<<<B_TOK / 256, 256, 0, stream>>>(eid, meta, pos, order);

    moe_gemm<D_DIM, true><<<dim3(H1_DIM / BN, MAX_TILES), 256, 0, stream>>>(
        xb, w1b, nullptr, hb, nullptr, meta, tileExp, tileRow, order, H1_DIM);
    moe_gemm<H1_DIM, false><<<dim3(O_DIM / BN, MAX_TILES), 256, 0, stream>>>(
        hb, w2b, b2, nullptr, y, meta, tileExp, tileRow, order, O_DIM);
}
